// TreeGNNRefinement_27238682591485
// MI455X (gfx1250) — matrix operand order repaired and checked by reading, not compiled, not run on hardware
//
#include <hip/hip_runtime.h>
#include <hip/hip_bf16.h>
#include <math.h>
#include <stdint.h>

// ---------------- types ----------------
typedef __attribute__((ext_vector_type(16))) __bf16 bf16x16;
typedef __attribute__((ext_vector_type(8)))  __bf16 bf16x8;
typedef __attribute__((ext_vector_type(8)))  float  v8f;

static constexpr int Bb  = 256;     // batch
static constexpr int Nn  = 128;     // seq len
static constexpr int Dmm = 256;     // model dim
static constexpr int NHh = 4;       // heads
static constexpr int HD  = 64;      // head dim
static constexpr int Mrows = Bb * Nn; // 32768

#define NEGV (-1e9f)

__device__ __forceinline__ float gelu_f(float x) {
  return 0.5f * x * (1.0f + erff(x * 0.70710678118654752f));
}
__device__ __forceinline__ float sigm_f(float x) {
  return 1.0f / (1.0f + __expf(-x));
}

// A-matrix 16x32 bf16 fragment (ISA 7.12.2): lane-half lh picks K-groups
// {0..7,16..23} (lh=0) or {8..15,24..31} (lh=1); row = lane&15 chosen by caller.
__device__ __forceinline__ bf16x16 fragA(const __bf16* rowp, int lh) {
  union { bf16x16 v; bf16x8 h[2]; } u;
  u.h[0] = *(const bf16x8*)(rowp + lh * 8);
  u.h[1] = *(const bf16x8*)(rowp + lh * 8 + 16);
  return u.v;
}
// B-matrix 32x16 bf16 fragment: lane-half lh picks K-groups {0..7,16..23} (lh=0)
// or {8..15,24..31} (lh=1); colp points at the 32 contiguous K-elements of column n (= lane&15).
__device__ __forceinline__ bf16x16 fragB(const __bf16* colp, int lh) {
  union { bf16x16 v; bf16x8 h[2]; } u;
  u.h[0] = *(const bf16x8*)(colp + lh * 8);
  u.h[1] = *(const bf16x8*)(colp + lh * 8 + 16);
  return u.v;
}
__device__ __forceinline__ v8f wmma_bf(bf16x16 a, bf16x16 b, v8f c) {
  return __builtin_amdgcn_wmma_f32_16x16x32_bf16(false, a, false, b, (short)0, c,
                                                 false, false);
}

// Async global->LDS 16-byte copy (CDNA5 §15.18.3 op 98) + wait (ASYNCcnt).
// lds_off = per-lane LDS byte offset (low 32 bits of flat shared address).
__device__ __forceinline__ void async_g2l_b128(uint32_t lds_off, const void* gptr,
                                               void* lds_fallback) {
#if defined(__HIP_DEVICE_COMPILE__)
  asm volatile("global_load_async_to_lds_b128 %0, %1, off"
               :: "v"(lds_off), "v"(gptr)
               : "memory");
  (void)lds_fallback;
#else
  *(bf16x8*)lds_fallback = *(const bf16x8*)gptr;
#endif
}
__device__ __forceinline__ void wait_async0() {
#if defined(__HIP_DEVICE_COMPILE__)
  asm volatile("s_wait_asynccnt 0x0" ::: "memory");
#endif
}

// ---------------- elementwise / conversion kernels ----------------
__global__ void copyf_k(const float* __restrict__ s, float* __restrict__ d, size_t n) {
  size_t i = (size_t)blockIdx.x * 256 + threadIdx.x;
  if (i < n) d[i] = s[i];
}
__global__ void zerof_k(float* __restrict__ p, size_t n) {
  size_t i = (size_t)blockIdx.x * 256 + threadIdx.x;
  if (i < n) p[i] = 0.0f;
}
__global__ void convcp_k(const float* __restrict__ s, __bf16* __restrict__ d, size_t n) {
  size_t i = (size_t)blockIdx.x * 256 + threadIdx.x;
  if (i < n) d[i] = (__bf16)s[i];
}
// per-slice transpose-convert: dst[sl][n*Kd+k] = bf16(src[sl][k*Nd+n])
__global__ void convtr_k(const float* __restrict__ src, __bf16* __restrict__ dst,
                         int Kd, int Nd) {
  size_t tot = (size_t)Kd * Nd;
  size_t sl = blockIdx.y;
  size_t i = (size_t)blockIdx.x * 256 + threadIdx.x;
  if (i < tot) {
    int n = (int)(i / Kd);
    int k = (int)(i % Kd);
    dst[sl * tot + i] = (__bf16)src[sl * tot + (size_t)k * Nd + n];
  }
}
// Hr f32 -> Hr bf16 and comb[:,0:256]
__global__ void hrpack_k(const float* __restrict__ Hr, __bf16* __restrict__ HrBf,
                         __bf16* __restrict__ comb) {
  int row = blockIdx.x, t = threadIdx.x;
  float v = Hr[(size_t)row * Dmm + t];
  HrBf[(size_t)row * Dmm + t] = (__bf16)v;
  comb[(size_t)row * 1024 + t] = (__bf16)v;
}
// scatter children onto heads (f32 global atomics)
__global__ void scatter_k(const float* __restrict__ Hr, const int* __restrict__ ph,
                          const float* __restrict__ msk, float* __restrict__ csum,
                          float* __restrict__ ccnt) {
  int row = blockIdx.x;
  int b = row / Nn;
  int hv = ph[row];
  hv = hv < 0 ? 0 : (hv > Nn - 1 ? Nn - 1 : hv);
  int dst = b * Nn + hv;
  atomicAdd(&csum[(size_t)dst * Dmm + threadIdx.x], Hr[(size_t)row * Dmm + threadIdx.x]);
  if (threadIdx.x == 0) atomicAdd(&ccnt[dst], msk[row]);
}
// Xcat = bf16([Hr, child_sum/max(cnt,1)])
__global__ void pack2_k(const float* __restrict__ Hr, const float* __restrict__ csum,
                        const float* __restrict__ ccnt, __bf16* __restrict__ Xcat) {
  int row = blockIdx.x, t = threadIdx.x;
  float cnt = ccnt[row];
  cnt = cnt > 1.0f ? cnt : 1.0f;
  Xcat[(size_t)row * 512 + t] = (__bf16)Hr[(size_t)row * Dmm + t];
  Xcat[(size_t)row * 512 + 256 + t] = (__bf16)(csum[(size_t)row * Dmm + t] / cnt);
}

// ---------------- generic bf16 WMMA GEMM ----------------
// Y[row*ldo + coloff + col] = act(X[M,Kd] @ Wt[n][k]^T + bias[col])
// grid = (Mrows/128, 256/64); 8 waves, each a 32x32 tile; no LDS needed.
template <int ACT, bool OBF>
__global__ __launch_bounds__(256) void gemm_k(const __bf16* __restrict__ X,
                                              const __bf16* __restrict__ Wt,
                                              const float* __restrict__ bias,
                                              void* __restrict__ Y, int Kd, int ldo,
                                              int coloff) {
  const int tid = threadIdx.x, lane = tid & 31, w = tid >> 5;
  const int l16 = lane & 15, lh = lane >> 4;
  const int m0 = blockIdx.x * 128 + (w >> 1) * 32;
  const int n0 = blockIdx.y * 64 + (w & 1) * 32;
  v8f c00 = {}, c01 = {}, c10 = {}, c11 = {};
  const __bf16* xr0 = X + (size_t)(m0 + l16) * Kd;
  const __bf16* xr1 = X + (size_t)(m0 + 16 + l16) * Kd;
  const __bf16* wr0 = Wt + (size_t)(n0 + l16) * Kd;
  const __bf16* wr1 = Wt + (size_t)(n0 + 16 + l16) * Kd;
  for (int kk = 0; kk < Kd; kk += 32) {
    if (kk + 32 < Kd) {  // software prefetch one K-step ahead (global_prefetch_b8)
      __builtin_prefetch(xr0 + kk + 32, 0, 0);
      __builtin_prefetch(wr0 + kk + 32, 0, 0);
    }
    bf16x16 a0 = fragA(xr0 + kk, lh);
    bf16x16 a1 = fragA(xr1 + kk, lh);
    bf16x16 b0 = fragB(wr0 + kk, lh);
    bf16x16 b1 = fragB(wr1 + kk, lh);
    c00 = wmma_bf(a0, b0, c00);
    c01 = wmma_bf(a0, b1, c01);
    c10 = wmma_bf(a1, b0, c10);
    c11 = wmma_bf(a1, b1, c11);
  }
  for (int im = 0; im < 2; ++im) {
    for (int jn = 0; jn < 2; ++jn) {
      v8f c = im ? (jn ? c11 : c10) : (jn ? c01 : c00);
      int col = n0 + jn * 16 + l16;
      float bv = bias ? bias[col] : 0.0f;
      for (int r = 0; r < 8; ++r) {
        int row = m0 + im * 16 + r + lh * 8;
        float v = c[r] + bv;
        if (ACT == 1) v = gelu_f(v);
        else if (ACT == 2) v = sigm_f(v);
        size_t o = (size_t)row * ldo + coloff + col;
        if (OBF) ((__bf16*)Y)[o] = (__bf16)v;
        else ((float*)Y)[o] = v;
      }
    }
  }
}

// ---------------- fused attention per (b,h) ----------------
// GATHER: K/V rows gathered through clipped pred_heads (head-attention)
// SIB:    sibling mask applied to scores
template <bool GATHER, bool SIB>
__global__ __launch_bounds__(256) void attn_k(const __bf16* __restrict__ Q,
                                              const __bf16* __restrict__ Kb,
                                              const __bf16* __restrict__ Vb,
                                              const int* __restrict__ ph,
                                              const float* __restrict__ msk,
                                              __bf16* __restrict__ AO) {
  __shared__ __bf16 Ks[Nn][72];          // K tile (gathered), async-staged
  __shared__ __bf16 VT[HD][Nn + 8];      // V^T staged (gathered if GATHER)
  __shared__ __bf16 P[8][16][Nn + 8];    // per-wave softmax strip
  __shared__ int sidx[Nn];
  __shared__ float smk[Nn];
  const int b = blockIdx.x >> 2;
  const int h = blockIdx.x & 3;
  const int tid = threadIdx.x, lane = tid & 31, w = tid >> 5;
  const int l16 = lane & 15, lh = lane >> 4;

  if (tid < Nn) {
    int hv = ph[b * Nn + tid];
    hv = hv < 0 ? 0 : (hv > Nn - 1 ? Nn - 1 : hv);
    sidx[tid] = hv;
    smk[tid] = msk ? msk[b * Nn + tid] : 1.0f;
  }
  __syncthreads();
  // K tile: 1024 x 16B async copies global->LDS (tracked by ASYNCcnt)
  for (int i = tid; i < Nn * 8; i += 256) {
    int pos = i >> 3;
    int ch = i & 7;
    int src = GATHER ? sidx[pos] : pos;
    const __bf16* gp = Kb + (size_t)(b * Nn + src) * Dmm + h * HD + ch * 8;
    __bf16* lp = &Ks[pos][ch * 8];
    async_g2l_b128((uint32_t)(uintptr_t)lp, gp, lp);
  }
  // V^T (transpose scatter; element-wise, regular path)
  for (int i = tid; i < HD * Nn; i += 256) {
    int pos = i >> 6;
    int d = i & 63;
    int src = GATHER ? sidx[pos] : pos;
    VT[d][pos] = Vb[(size_t)(b * Nn + src) * Dmm + h * HD + d];
  }
  wait_async0();
  __syncthreads();

  const int q0 = w * 16;
  const __bf16* qrow = Q + (size_t)(b * Nn + q0 + l16) * Dmm + h * HD;
  bf16x16 aq0 = fragA(qrow, lh);
  bf16x16 aq1 = fragA(qrow + 32, lh);

  v8f S[8];
  for (int j = 0; j < 8; ++j) {
    v8f c = {};
    const __bf16* krow = &Ks[j * 16 + l16][0];
    c = wmma_bf(aq0, fragB(krow, lh), c);
    c = wmma_bf(aq1, fragB(krow + 32, lh), c);
    S[j] = c;
  }

  const float scale = 0.125f;  // 1/sqrt(64)
  float mx[8], sm[8];
  for (int r = 0; r < 8; ++r) mx[r] = -3.0e38f;
  for (int j = 0; j < 8; ++j) {
    int k = j * 16 + l16;
    for (int r = 0; r < 8; ++r) {
      int q = q0 + r + lh * 8;
      float v = S[j][r] * scale;
      if (SIB) {
        bool ok = (sidx[q] == sidx[k]) && (q != k) && (smk[q] > 0.0f) && (smk[k] > 0.0f);
        if (!ok) v = NEGV;
      }
      S[j][r] = v;
      mx[r] = fmaxf(mx[r], v);
    }
  }
  for (int r = 0; r < 8; ++r) {
    for (int off = 8; off >= 1; off >>= 1) mx[r] = fmaxf(mx[r], __shfl_xor(mx[r], off, 32));
    sm[r] = 0.0f;
  }
  for (int j = 0; j < 8; ++j)
    for (int r = 0; r < 8; ++r) {
      float e = __expf(S[j][r] - mx[r]);
      S[j][r] = e;
      sm[r] += e;
    }
  for (int r = 0; r < 8; ++r) {
    for (int off = 8; off >= 1; off >>= 1) sm[r] += __shfl_xor(sm[r], off, 32);
    sm[r] = 1.0f / sm[r];
  }
  // relayout C->A through LDS (same-wave; compiler inserts s_wait_dscnt)
  for (int j = 0; j < 8; ++j)
    for (int r = 0; r < 8; ++r)
      P[w][r + lh * 8][j * 16 + l16] = (__bf16)(S[j][r] * sm[r]);

  v8f O[4] = {};
  for (int kc = 0; kc < 4; ++kc) {
    bf16x16 ap = fragA(&P[w][l16][kc * 32], lh);
    for (int c2 = 0; c2 < 4; ++c2)
      O[c2] = wmma_bf(ap, fragB(&VT[c2 * 16 + l16][kc * 32], lh), O[c2]);
  }
  for (int c2 = 0; c2 < 4; ++c2)
    for (int r = 0; r < 8; ++r) {
      int q = q0 + r + lh * 8;
      AO[(size_t)(b * Nn + q) * Dmm + h * HD + c2 * 16 + l16] = (__bf16)O[c2][r];
    }
}

// ---------------- residual + gated fuse + LayerNorm ----------------
__global__ __launch_bounds__(256) void ln_k(float* __restrict__ Hr,
                                            const float* __restrict__ g,
                                            const float* __restrict__ u,
                                            const float* __restrict__ lg,
                                            const float* __restrict__ lb) {
  __shared__ float red[8];
  int row = blockIdx.x, t = threadIdx.x, lane = t & 31, w = t >> 5;
  size_t o = (size_t)row * Dmm + t;
  float hn = Hr[o] + g[o] * u[o];
  float s = hn;
  for (int off = 16; off >= 1; off >>= 1) s += __shfl_xor(s, off, 32);
  if (lane == 0) red[w] = s;
  __syncthreads();
  float tot = 0.0f;
  for (int i = 0; i < 8; ++i) tot += red[i];
  float mu = tot * (1.0f / Dmm);
  float dv = hn - mu;
  __syncthreads();
  float s2 = dv * dv;
  for (int off = 16; off >= 1; off >>= 1) s2 += __shfl_xor(s2, off, 32);
  if (lane == 0) red[w] = s2;
  __syncthreads();
  float tot2 = 0.0f;
  for (int i = 0; i < 8; ++i) tot2 += red[i];
  float var = tot2 * (1.0f / Dmm);
  Hr[o] = dv * rsqrtf(var + 1e-5f) * lg[t] + lb[t];
}

// out[row] = dot(X[row,:], wv) + b
__global__ __launch_bounds__(256) void rowdot_k(const __bf16* __restrict__ X,
                                                const float* __restrict__ wv,
                                                const float* __restrict__ bsc,
                                                float* __restrict__ out) {
  __shared__ float red[8];
  int row = blockIdx.x, t = threadIdx.x, lane = t & 31, w = t >> 5;
  float s = (float)X[(size_t)row * Dmm + t] * wv[t];
  for (int off = 16; off >= 1; off >>= 1) s += __shfl_xor(s, off, 32);
  if (lane == 0) red[w] = s;
  __syncthreads();
  if (t == 0) {
    float tot = 0.0f;
    for (int i = 0; i < 8; ++i) tot += red[i];
    out[row] = tot + bsc[0];
  }
}

// scores[b,q,m] = Hh[b,q,:].Hb[b,m,:] + hdb[b,q] + hhb[b,m]; one block per b.
__global__ __launch_bounds__(256) void scores_k(const __bf16* __restrict__ Hh,
                                                const __bf16* __restrict__ Hb,
                                                const float* __restrict__ hdb,
                                                const float* __restrict__ hhb,
                                                float* __restrict__ out) {
  int b = blockIdx.x, tid = threadIdx.x, lane = tid & 31, w = tid >> 5;
  int l16 = lane & 15, lh = lane >> 4;
  int q0 = w * 16;
  const __bf16* hrow = Hh + (size_t)(b * Nn + q0 + l16) * Dmm;
  v8f C[8] = {};
  for (int kk = 0; kk < Dmm; kk += 32) {
    bf16x16 a = fragA(hrow + kk, lh);
    for (int j = 0; j < 8; ++j) {
      const __bf16* mrow = Hb + (size_t)(b * Nn + j * 16 + l16) * Dmm + kk;
      C[j] = wmma_bf(a, fragB(mrow, lh), C[j]);
    }
  }
  for (int j = 0; j < 8; ++j)
    for (int r = 0; r < 8; ++r) {
      int q = q0 + r + lh * 8;
      int m2 = j * 16 + l16;
      out[((size_t)b * Nn + q) * Nn + m2] = C[j][r] + hdb[b * Nn + q] + hhb[b * Nn + m2];
    }
}

// ---------------- host ----------------
extern "C" void kernel_launch(void* const* d_in, const int* in_sizes, int n_in,
                              void* d_out, int out_size, void* d_ws, size_t ws_size,
                              hipStream_t stream) {
  const float* H      = (const float*)d_in[0];
  const int*   ph     = (const int*)d_in[1];
  const float* maskp  = (const float*)d_in[2];
  const float* ha_w   = (const float*)d_in[3];
  const float* ha_b   = (const float*)d_in[4];
  const float* sa_w   = (const float*)d_in[5];
  const float* sa_b   = (const float*)d_in[6];
  const float* ct_w   = (const float*)d_in[7];
  const float* ct_b   = (const float*)d_in[8];
  const float* gate_w = (const float*)d_in[9];
  const float* gate_b = (const float*)d_in[10];
  const float* tr_w   = (const float*)d_in[11];
  const float* tr_b   = (const float*)d_in[12];
  const float* ln_g   = (const float*)d_in[13];
  const float* ln_b   = (const float*)d_in[14];
  const float* arcw_h = (const float*)d_in[15];
  const float* arcb_h = (const float*)d_in[16];
  const float* arcw_d = (const float*)d_in[17];
  const float* arcb_d = (const float*)d_in[18];
  const float* bil    = (const float*)d_in[19];
  const float* bh_w   = (const float*)d_in[20];
  const float* bh_b   = (const float*)d_in[21];
  const float* bd_w   = (const float*)d_in[22];
  const float* bd_b   = (const float*)d_in[23];

  char* ws = (char*)d_ws;
  size_t off = 0;
  auto arena = [&](size_t bytes) -> void* {
    off = (off + 255) & ~(size_t)255;
    void* p = ws + off;
    off += bytes;
    return p;
  };
  const size_t DD = (size_t)Dmm * Dmm;  // 65536
  __bf16* haT   = (__bf16*)arena(3 * 4 * DD * 2);
  __bf16* saT   = (__bf16*)arena(3 * 4 * DD * 2);
  __bf16* ctT   = (__bf16*)arena(3 * 512 * 256 * 2);
  __bf16* gateT = (__bf16*)arena(3 * 1024 * 256 * 2);
  __bf16* trT   = (__bf16*)arena(3 * 1024 * 256 * 2);
  __bf16* archT = (__bf16*)arena(DD * 2);
  __bf16* arcdT = (__bf16*)arena(DD * 2);
  __bf16* bilT  = (__bf16*)arena(DD * 2);
  float*  Hr    = (float*)arena((size_t)Mrows * Dmm * 4);
  __bf16* HrBf  = (__bf16*)arena((size_t)Mrows * Dmm * 2);
  __bf16* Qb    = (__bf16*)arena((size_t)Mrows * Dmm * 2);
  __bf16* Kb    = (__bf16*)arena((size_t)Mrows * Dmm * 2);
  __bf16* Vb    = (__bf16*)arena((size_t)Mrows * Dmm * 2);
  __bf16* AO    = (__bf16*)arena((size_t)Mrows * Dmm * 2);
  float*  csum  = (float*)arena((size_t)Mrows * Dmm * 4);
  float*  ccnt  = (float*)arena((size_t)Mrows * 4);
  __bf16* Xcat  = (__bf16*)arena((size_t)Mrows * 512 * 2);
  __bf16* comb  = (__bf16*)arena((size_t)Mrows * 1024 * 2);
  // reuse (disjoint in time):
  float* gbuf = csum;            // gate output (after scatter/pack done)
  float* ubuf = (float*)Xcat;    // tr output (after ct GEMM done)
  __bf16* Hh = Qb;               // final phase
  __bf16* Hd = Kb;
  __bf16* Hb = Vb;
  float* hhb = csum;             // final phase
  float* hdb = csum + Mrows;

  dim3 blk(256);
  const size_t MD = (size_t)Mrows * Dmm;

  // ---- one-time (per call) weight convert+transpose ----
  convtr_k<<<dim3((unsigned)((DD + 255) / 256), 12), blk, 0, stream>>>(ha_w, haT, 256, 256);
  convtr_k<<<dim3((unsigned)((DD + 255) / 256), 12), blk, 0, stream>>>(sa_w, saT, 256, 256);
  convtr_k<<<dim3((unsigned)((512 * 256 + 255) / 256), 3), blk, 0, stream>>>(ct_w, ctT, 512, 256);
  convtr_k<<<dim3((unsigned)((1024 * 256 + 255) / 256), 3), blk, 0, stream>>>(gate_w, gateT, 1024, 256);
  convtr_k<<<dim3((unsigned)((1024 * 256 + 255) / 256), 3), blk, 0, stream>>>(tr_w, trT, 1024, 256);
  convtr_k<<<dim3((unsigned)((DD + 255) / 256), 1), blk, 0, stream>>>(arcw_h, archT, 256, 256);
  convtr_k<<<dim3((unsigned)((DD + 255) / 256), 1), blk, 0, stream>>>(arcw_d, arcdT, 256, 256);
  convcp_k<<<(unsigned)((DD + 255) / 256), blk, 0, stream>>>(bil, bilT, DD);  // Wt == bilinear

  auto gemm = [&](const __bf16* X, const __bf16* Wt, const float* bias, void* Y,
                  int Kd, int act, bool obf, int ldo, int coloff) {
    dim3 g(Mrows / 128, Dmm / 64);
    if (obf) {
      if (act == 0)      gemm_k<0, true><<<g, blk, 0, stream>>>(X, Wt, bias, Y, Kd, ldo, coloff);
      else if (act == 1) gemm_k<1, true><<<g, blk, 0, stream>>>(X, Wt, bias, Y, Kd, ldo, coloff);
      else               gemm_k<2, true><<<g, blk, 0, stream>>>(X, Wt, bias, Y, Kd, ldo, coloff);
    } else {
      if (act == 0)      gemm_k<0, false><<<g, blk, 0, stream>>>(X, Wt, bias, Y, Kd, ldo, coloff);
      else if (act == 1) gemm_k<1, false><<<g, blk, 0, stream>>>(X, Wt, bias, Y, Kd, ldo, coloff);
      else               gemm_k<2, false><<<g, blk, 0, stream>>>(X, Wt, bias, Y, Kd, ldo, coloff);
    }
  };

  copyf_k<<<(unsigned)((MD + 255) / 256), blk, 0, stream>>>(H, Hr, MD);

  for (int i = 0; i < 3; ++i) {
    hrpack_k<<<Mrows, blk, 0, stream>>>(Hr, HrBf, comb);  // comb[:,0:256] = Hr
    const __bf16* haTi = haT + (size_t)i * 4 * DD;
    const float* habi = ha_b + (size_t)i * 4 * 256;
    // head attention: project Hr, gather K/V in-attention (gather commutes with proj)
    gemm(HrBf, haTi + 0 * DD, habi + 0,   Qb, 256, 0, true, 256, 0);
    gemm(HrBf, haTi + 1 * DD, habi + 256, Kb, 256, 0, true, 256, 0);
    gemm(HrBf, haTi + 2 * DD, habi + 512, Vb, 256, 0, true, 256, 0);
    attn_k<true, false><<<Bb * NHh, blk, 0, stream>>>(Qb, Kb, Vb, ph, nullptr, AO);
    gemm(AO, haTi + 3 * DD, habi + 768, comb, 256, 0, true, 1024, 256);  // head_msg
    // child path
    zerof_k<<<(unsigned)((MD + 255) / 256), blk, 0, stream>>>(csum, MD);
    zerof_k<<<(unsigned)((Mrows + 255) / 256), blk, 0, stream>>>(ccnt, (size_t)Mrows);
    scatter_k<<<Mrows, blk, 0, stream>>>(Hr, ph, maskp, csum, ccnt);
    pack2_k<<<Mrows, blk, 0, stream>>>(Hr, csum, ccnt, Xcat);
    gemm(Xcat, ctT + (size_t)i * 512 * 256, ct_b + i * 256, comb, 512, 1, true, 1024, 512);
    // sibling attention
    const __bf16* saTi = saT + (size_t)i * 4 * DD;
    const float* sabi = sa_b + (size_t)i * 4 * 256;
    gemm(HrBf, saTi + 0 * DD, sabi + 0,   Qb, 256, 0, true, 256, 0);
    gemm(HrBf, saTi + 1 * DD, sabi + 256, Kb, 256, 0, true, 256, 0);
    gemm(HrBf, saTi + 2 * DD, sabi + 512, Vb, 256, 0, true, 256, 0);
    attn_k<false, true><<<Bb * NHh, blk, 0, stream>>>(Qb, Kb, Vb, ph, maskp, AO);
    gemm(AO, saTi + 3 * DD, sabi + 768, comb, 256, 0, true, 1024, 768);  // sib_msg
    // gated fuse + residual + LN
    gemm(comb, gateT + (size_t)i * 1024 * 256, gate_b + i * 256, gbuf, 1024, 2, false, 256, 0);
    gemm(comb, trT + (size_t)i * 1024 * 256, tr_b + i * 256, ubuf, 1024, 1, false, 256, 0);
    ln_k<<<Mrows, blk, 0, stream>>>(Hr, gbuf, ubuf, ln_g + i * 256, ln_b + i * 256);
  }

  // ---- biaffine scoring ----
  convcp_k<<<(unsigned)((MD + 255) / 256), blk, 0, stream>>>(Hr, HrBf, MD);
  gemm(HrBf, archT, arcb_h, Hh, 256, 0, true, 256, 0);
  gemm(HrBf, arcdT, arcb_d, Hd, 256, 0, true, 256, 0);
  gemm(Hd, bilT, nullptr, Hb, 256, 0, true, 256, 0);  // hb = h_dep @ bilinear^T
  rowdot_k<<<Mrows, blk, 0, stream>>>(Hh, bh_w, bh_b, hhb);
  rowdot_k<<<Mrows, blk, 0, stream>>>(Hd, bd_w, bd_b, hdb);
  scores_k<<<Bb, blk, 0, stream>>>(Hh, Hb, hdb, hhb, (float*)d_out + MD);
  copyf_k<<<(unsigned)((MD + 255) / 256), blk, 0, stream>>>(Hr, (float*)d_out, MD);
}
